// TreeLSTMEncoder_22849226014993
// MI455X (gfx1250) — matrix operand order repaired and checked by reading, not compiled, not run on hardware
//
#include <hip/hip_runtime.h>
#include <hip/hip_bf16.h>
#include <stdint.h>

// Problem constants (match reference)
#define B_  256
#define N_  256
#define K_  3
#define V_  32000
#define E_  256
#define H_  256
#define G_  1536      // (3+K)*H
#define KK_ 1024      // E + K*H : fused contraction dim

typedef __attribute__((ext_vector_type(16))) __bf16 v16bf;
typedef __attribute__((ext_vector_type(8)))  float  v8f;

union FragBF {
    uint4 q[2];
    v16bf v;
};

__device__ __forceinline__ unsigned short f2bf(float f) {
    unsigned int u = __float_as_uint(f);
    unsigned int r = u + 0x7FFFu + ((u >> 16) & 1u);   // round-to-nearest-even
    return (unsigned short)(r >> 16);
}

// ---- Pack W_t[g][kk] (bf16, K-contiguous): kk<E -> Wx_w[kk][g], else Uh_w[kk-E][g]
//      Also clears the 256-half zero pad used for masked child rows.
__global__ __launch_bounds__(256)
void pack_weights(const float* __restrict__ Wx_w,
                  const float* __restrict__ Uh_w,
                  unsigned short* __restrict__ Wt,
                  unsigned short* __restrict__ zpad) {
    int tid = blockIdx.x * blockDim.x + threadIdx.x;   // over G_*KK_
    int kk  = tid & (KK_ - 1);
    int g   = tid >> 10;
    float v = (kk < E_) ? Wx_w[(size_t)kk * G_ + g]
                        : Uh_w[(size_t)(kk - E_) * G_ + g];
    Wt[tid] = f2bf(v);
    if (tid < 256) zpad[tid] = 0;
}

__global__ __launch_bounds__(256)
void pack_emb(const float* __restrict__ emb, unsigned short* __restrict__ ebf) {
    int tid = blockIdx.x * blockDim.x + threadIdx.x;   // over V_*E_
    ebf[tid] = f2bf(emb[tid]);
}

// ---- Per-step gate GEMM: gates[b,g] = [x_t | m*h_children] @ [Wx_w;Uh_w] + Wx_b + m@Uh_b
// Block: 256 thr = 8 waves. Tile M=64 (rows b), N=128 (cols g). Wave: 16x64.
__global__ __launch_bounds__(256)
void tree_gates_gemm(const int* __restrict__ node_ids,          // [B,N]
                     const int* __restrict__ children,          // [B,N,K]
                     const unsigned short* __restrict__ ebf,    // [V,E]  bf16
                     const unsigned short* __restrict__ hbf,    // [B,N,H] bf16
                     const unsigned short* __restrict__ Wt,     // [G,KK] bf16 (K-contig)
                     const unsigned short* __restrict__ zpad,   // [256] zeros
                     const float* __restrict__ Wx_b,            // [G]
                     const float* __restrict__ Uh_b,            // [K,G]
                     float* __restrict__ gates,                 // [B,G]
                     int t) {
    __shared__ int rowoff[64][4];   // [row][0]=emb offset, [1..3]=child h offset or -1

    const int lane = threadIdx.x & 31;
    const int wave = threadIdx.x >> 5;
    const int mw   = wave & 3;                 // 4 M-tiles of 16
    const int nw   = wave >> 2;                // 2 N-groups of 64
    const int b_base   = blockIdx.y * 64;
    const int col_base = blockIdx.x * 128 + nw * 64;

    if (threadIdx.x < 64) {
        int b = b_base + threadIdx.x;
        rowoff[threadIdx.x][0] = node_ids[b * N_ + t] * E_;
        #pragma unroll
        for (int k = 0; k < K_; ++k) {
            int raw = children[((size_t)b * N_ + t) * K_ + k];
            rowoff[threadIdx.x][k + 1] = (raw < t) ? (b * N_ + raw) * H_ : -1;
        }
    }
    __syncthreads();

    const int r     = mw * 16 + (lane & 15);   // A row within block tile
    const int koff  = (lane >> 4) * 8;         // A-lane K offset (ISA 16-bit A layout)
    const int kboff = (lane >> 4) * 8;         // B-lane K offset (same K order as A)
    const int gcol  = lane & 15;

    // Hoist all source pointers out of the K loop (branchless hot loop).
    const unsigned short* abase[4];
    {
        abase[0] = ebf + rowoff[r][0] + koff;
        #pragma unroll
        for (int k = 1; k < 4; ++k) {
            int rk = rowoff[r][k];
            abase[k] = ((rk >= 0) ? (hbf + rk) : zpad) + koff;  // masked -> zero pad
        }
    }
    const unsigned short* bbase[4];
    #pragma unroll
    for (int i = 0; i < 4; ++i)
        bbase[i] = Wt + (size_t)(col_base + i * 16 + gcol) * KK_ + kboff;

    v8f acc[4] = {};

    #pragma unroll
    for (int kc = 0; kc < 4; ++kc) {           // chunk source is constant per kc
        const unsigned short* ab = abase[kc];
        #pragma unroll 4
        for (int j = 0; j < 8; ++j) {          // 8 x K=32 per chunk
            const int kk0 = kc * 256 + j * 32;
            FragBF a;
            a.q[0] = *(const uint4*)(ab + j * 32);        // K = kk0+koff .. +7
            a.q[1] = *(const uint4*)(ab + j * 32 + 16);   // K = kk0+16+koff .. +23
            __builtin_prefetch((const void*)(bbase[0] + kk0 + 256), 0, 1);
            #pragma unroll
            for (int i = 0; i < 4; ++i) {
                FragBF bm;
                bm.q[0] = *(const uint4*)(bbase[i] + kk0);
                bm.q[1] = *(const uint4*)(bbase[i] + kk0 + 16);
                acc[i] = __builtin_amdgcn_wmma_f32_16x16x32_bf16(
                    false, a.v, false, bm.v, (short)0, acc[i], false, false);
            }
        }
    }

    // Epilogue: + Wx_b + sum_k m[b,k]*Uh_b[k], store gates (C layout: M=e+8*(lane>=16), N=lane%16)
    const int erow = (lane >> 4) << 3;
    #pragma unroll
    for (int i = 0; i < 4; ++i) {
        int g = col_base + i * 16 + gcol;
        float bias0 = Wx_b[g];
        #pragma unroll
        for (int e = 0; e < 8; ++e) {
            int rr = mw * 16 + e + erow;
            float v = acc[i][e] + bias0;
            if (rowoff[rr][1] >= 0) v += Uh_b[0 * G_ + g];
            if (rowoff[rr][2] >= 0) v += Uh_b[1 * G_ + g];
            if (rowoff[rr][3] >= 0) v += Uh_b[2 * G_ + g];
            gates[(size_t)(b_base + rr) * G_ + g] = v;
        }
    }
}

// ---- Per-step LSTM cell: c = sig(i)*tanh(u) + sum_k m*sig(f_k)*c_child ; h = sig(o)*tanh(c)
__global__ __launch_bounds__(256)
void tree_update(const int* __restrict__ children,      // [B,N,K]
                 const float* __restrict__ gates,       // [B,G]
                 float* __restrict__ cbuf,              // [B,N,H] f32
                 unsigned short* __restrict__ hbf,      // [B,N,H] bf16
                 float* __restrict__ out,               // [B,H]
                 int t) {
    int idx = blockIdx.x * blockDim.x + threadIdx.x;    // B_*H_
    int b = idx >> 8;
    int h = idx & (H_ - 1);
    const float* gr = gates + (size_t)b * G_;

    float i_g = 1.0f / (1.0f + __expf(-gr[h]));
    float o_g = 1.0f / (1.0f + __expf(-gr[H_ + h]));
    float u_g = tanhf(gr[2 * H_ + h]);
    float c = i_g * u_g;

    #pragma unroll
    for (int k = 0; k < K_; ++k) {
        int raw = children[((size_t)b * N_ + t) * K_ + k];
        if (raw < t) {
            float f = 1.0f / (1.0f + __expf(-gr[3 * H_ + k * H_ + h]));
            c += f * cbuf[((size_t)b * N_ + raw) * H_ + h];
        }
    }
    float hv = o_g * tanhf(c);
    cbuf[((size_t)b * N_ + t) * H_ + h] = c;
    hbf [((size_t)b * N_ + t) * H_ + h] = f2bf(hv);
    if (t == N_ - 1) out[b * H_ + h] = hv;
}

extern "C" void kernel_launch(void* const* d_in, const int* in_sizes, int n_in,
                              void* d_out, int out_size, void* d_ws, size_t ws_size,
                              hipStream_t stream) {
    (void)in_sizes; (void)n_in; (void)out_size; (void)ws_size;

    const int*   node_ids = (const int*)  d_in[0];   // [B,N]
    const int*   children = (const int*)  d_in[1];   // [B,N,K]
    const float* emb      = (const float*)d_in[2];   // [V,E]
    const float* Wx_w     = (const float*)d_in[3];   // [E,G]
    const float* Wx_b     = (const float*)d_in[4];   // [G]
    const float* Uh_w     = (const float*)d_in[5];   // [K,H,G]
    const float* Uh_b     = (const float*)d_in[6];   // [K,G]
    float* out = (float*)d_out;                      // [B,H] f32

    // Workspace carve-up (all chunk sizes are multiples of 256B)
    char* p = (char*)d_ws;
    unsigned short* Wt   = (unsigned short*)p;  p += (size_t)G_ * KK_ * 2;        //  3.0 MB
    unsigned short* ebf  = (unsigned short*)p;  p += (size_t)V_ * E_  * 2;        // 15.6 MB
    unsigned short* hbf  = (unsigned short*)p;  p += (size_t)B_ * N_ * H_ * 2;    // 32.0 MB
    float*          cbuf = (float*)p;           p += (size_t)B_ * N_ * H_ * 4;    // 64.0 MB
    float*          gts  = (float*)p;           p += (size_t)B_ * G_ * 4;         //  1.5 MB
    unsigned short* zpad = (unsigned short*)p;                                    //  512 B

    hipLaunchKernelGGL(pack_weights, dim3((G_ * KK_) / 256), dim3(256), 0, stream,
                       Wx_w, Uh_w, Wt, zpad);
    hipLaunchKernelGGL(pack_emb, dim3((V_ * E_) / 256), dim3(256), 0, stream,
                       emb, ebf);

    for (int t = 0; t < N_; ++t) {
        hipLaunchKernelGGL(tree_gates_gemm, dim3(G_ / 128, B_ / 64), dim3(256), 0, stream,
                           node_ids, children, ebf, hbf, Wt, zpad, Wx_b, Uh_b, gts, t);
        hipLaunchKernelGGL(tree_update, dim3((B_ * H_) / 256), dim3(256), 0, stream,
                           children, gts, cbuf, hbf, out, t);
    }
}
